// CustomS4Model_13572096655882
// MI455X (gfx1250) — matrix operand order repaired and checked by reading, not compiled, not run on hardware
//
#include <hip/hip_runtime.h>
#include <math.h>

// Problem constants (match reference)
#define B_  32
#define L_  1024
#define H_  512
#define N_  64
#define NL_ 4
#define T_  32            // chunk length (= WMMA K depth, no padding)
#define C_  32            // chunks per sequence = L/T
#define SROW 132          // S row stride (floats): 128 data + 4 pad (bank spread)
#define BHL ((size_t)B_ * H_ * L_)

typedef __attribute__((ext_vector_type(16))) _Float16 v16h;
typedef __attribute__((ext_vector_type(8)))  float    v8f;

__device__ __forceinline__ float gelu_tanh(float x) {
    const float k0 = 0.7978845608028654f;   // sqrt(2/pi)
    const float k1 = 0.044715f;
    float x3 = x * x * x;
    return 0.5f * x * (1.0f + tanhf(k0 * (x + k1 * x3)));
}

// ---------------------------------------------------------------------------
// Embedding gather with transpose to channels-first [B, H, L] layout.
// ---------------------------------------------------------------------------
__global__ __launch_bounds__(256)
void embed_kernel(const int* __restrict__ x, const float* __restrict__ emb,
                  float* __restrict__ out) {
    __shared__ float tile[32][33];
    int blk   = blockIdx.x;          // B * (L/32) blocks
    int b     = blk >> 5;
    int ltile = blk & 31;
    int l0    = ltile * 32;
    int hin   = threadIdx.x & 31;
    int rowg  = threadIdx.x >> 5;    // 0..7

    for (int ht = 0; ht < H_ / 32; ++ht) {
        int h0 = ht * 32;
        for (int r = 0; r < 4; ++r) {
            int l   = rowg + 8 * r;
            int idx = x[b * L_ + l0 + l];
            tile[l][hin] = emb[(size_t)idx * H_ + h0 + hin];
        }
        __syncthreads();
        int lout = threadIdx.x & 31;
        for (int r = 0; r < 4; ++r) {
            int h = rowg + 8 * r;
            out[((size_t)b * H_ + h0 + h) * L_ + l0 + lout] = tile[lout][h];
        }
        __syncthreads();
    }
}

// ---------------------------------------------------------------------------
// One S4D layer via chunked scan, WMMA f16 16x16x32 with FULL K utilization.
// grid = (H, 4); block = 128 (4 waves). blockIdx.x = channel h,
// blockIdx.y = batch group of 8. Wave decode: ct = wid>>1 (col-tile of 16
// chunks), sub = wid&1 (row half).
//
// Phase A: A_big[160 x 32] @ U[32 x 32chunks], K = 32 exactly:
//   rows   0..31  : Toeplitz M[i][j] = K[i-j]          -> y_intra (2 tiles)
//   rows  32..95  : Re(lambda_n^{31-j})                -> Re(S_in) (4 tiles)
//   rows  96..159 : Im(lambda_n^{31-j})                -> Im(S_in) (4 tiles)
// Wave(sub=0): {M rows 0-15,  Re(P) tiles};  wave(sub=1): {M rows 16-31, Im(P)}
// -> each wave's intra tile is exactly its phase-B output row-half, so the
//    accumulator stays in registers as the phase-B C operand.
// Phase B: A2[32 x 128] = [Re(c*l^{i+1}) | -Im(c*l^{i+1})] @ S_stack[128 x 32].
//
// S transposed in LDS: S[chunk][0..63]=Re, [64..127]=Im, stride 132 floats.
// ---------------------------------------------------------------------------
__global__ __launch_bounds__(128)
void s4d_kernel(const float* __restrict__ u,      // [B,H,L]
                float* __restrict__ y,            // [B,H,L]
                const float* __restrict__ log_dt, // [H]   (layer slice)
                const float* __restrict__ A_log,  // [H,N]
                const float* __restrict__ A_imag, // [H,N]
                const float* __restrict__ C_re,   // [H,N]
                const float* __restrict__ C_im,   // [H,N]
                const float* __restrict__ Dp) {   // [H]
    const int h     = blockIdx.x;
    const int tid   = threadIdx.x;
    const int wid   = tid >> 5;
    const int lane  = tid & 31;
    const int m16   = lane & 15;   // row (A) or column (B/C/D) within tile
    const int khalf = lane >> 4;
    const int sub   = wid & 1;     // output row half (0: rows 0-15, 1: 16-31)
    const int ct    = wid >> 1;    // col-tile (16 chunks)

    __shared__ float lpow_re[T_ + 1][N_];   // lambda^d, d = 0..32
    __shared__ float lpow_im[T_ + 1][N_];
    __shared__ float chat_re[N_];           // 2*C*(lambda-1)/A
    __shared__ float chat_im[N_];
    __shared__ float Kcoef[T_];             // K[d] = Re sum_n chat*lambda^d
    __shared__ float S[C_][SROW];           // transposed S_in -> S_state

    // ---- per-channel setup: discretization + lambda powers -------------
    if (tid < N_) {
        int n = tid;
        float dt   = __expf(log_dt[h]);
        float a_re = -__expf(A_log[h * N_ + n]);
        float a_im = A_imag[h * N_ + n];
        float dre = dt * a_re, dim = dt * a_im;
        float er  = __expf(dre);
        float lr1 = er * __cosf(dim);
        float li1 = er * __sinf(dim);
        float lr = 1.f, li = 0.f;
        lpow_re[0][n] = 1.f; lpow_im[0][n] = 0.f;
        for (int d = 1; d <= T_; ++d) {
            float nr = lr * lr1 - li * li1;
            float ni = lr * li1 + li * lr1;
            lr = nr; li = ni;
            lpow_re[d][n] = lr; lpow_im[d][n] = li;
        }
        // chat = 2 * C * (exp(dtA) - 1) / A     (A undiscretized)
        float numr = lr1 - 1.f, numi = li1;
        float den  = a_re * a_re + a_im * a_im;
        float tr = (numr * a_re + numi * a_im) / den;
        float ti = (numi * a_re - numr * a_im) / den;
        float cr = C_re[h * N_ + n], ci = C_im[h * N_ + n];
        chat_re[n] = 2.f * (cr * tr - ci * ti);
        chat_im[n] = 2.f * (cr * ti + ci * tr);
    }
    __syncthreads();
    if (tid < T_) {
        int d = tid; float s = 0.f;
        for (int n = 0; n < N_; ++n)
            s += chat_re[n] * lpow_re[d][n] - chat_im[n] * lpow_im[d][n];
        Kcoef[d] = s;
    }
    __syncthreads();

    // ---- build loop-invariant A fragments (per wave, in registers) -----
    // A fragment order used here: lane m16 = row; element e -> k = 16*khalf + e
    const int irow = sub * 16 + m16;        // this wave's global output row
    v16h afrag[5];
    for (int t = 0; t < 5; ++t) {
        for (int e = 0; e < 16; ++e) {
            int k = khalf * 16 + e;   // 0..31, full K
            float v;
            if (t == 0) {                     // Toeplitz intra tile
                v = (k <= irow) ? Kcoef[irow - k] : 0.f;
            } else {                          // Vandermonde P tiles
                int n = (t - 1) * 16 + m16;
                v = sub ? lpow_im[31 - k][n] : lpow_re[31 - k][n];
            }
            afrag[t][e] = (_Float16)v;
        }
    }
    v16h a2[4];                               // phase-B A, K-steps of 32
    for (int q = 0; q < 4; ++q) {
        for (int e = 0; e < 16; ++e) {
            int kl = khalf * 16 + e;
            int kk = 32 * q + kl;
            float v;
            if (kk < 64) {
                int n = kk;
                v = chat_re[n] * lpow_re[irow + 1][n] -
                    chat_im[n] * lpow_im[irow + 1][n];
            } else {
                int n = kk - 64;
                v = -(chat_re[n] * lpow_im[irow + 1][n] +
                      chat_im[n] * lpow_re[irow + 1][n]);
            }
            a2[q][e] = (_Float16)v;
        }
    }

    const float Dh    = Dp[h];
    const int   chunk = ct * 16 + m16;      // this lane's column = chunk index
    const int   b_lo  = blockIdx.y * (B_ / 4);
    const int   b_hi  = b_lo + (B_ / 4);

    for (int b = b_lo; b < b_hi; ++b) {
        const float* ub = u + ((size_t)b * H_ + h) * L_;
        if (b + 1 < b_hi)                    // hide next row's latency
            __builtin_prefetch(u + ((size_t)(b + 1) * H_ + h) * L_, 0, 1);

        // ---- phase A B-fragment: 16 contiguous floats/lane, no masking --
        // B layout: col = m16, element e -> k = 16*khalf + e  (K = 32 full)
        const float4* uc4 = (const float4*)(ub + chunk * T_ + 16 * khalf);
        float4 u0 = uc4[0], u1 = uc4[1], u2 = uc4[2], u3 = uc4[3];
        v16h bfrag;
        bfrag[0]  = (_Float16)u0.x; bfrag[1]  = (_Float16)u0.y;
        bfrag[2]  = (_Float16)u0.z; bfrag[3]  = (_Float16)u0.w;
        bfrag[4]  = (_Float16)u1.x; bfrag[5]  = (_Float16)u1.y;
        bfrag[6]  = (_Float16)u1.z; bfrag[7]  = (_Float16)u1.w;
        bfrag[8]  = (_Float16)u2.x; bfrag[9]  = (_Float16)u2.y;
        bfrag[10] = (_Float16)u2.z; bfrag[11] = (_Float16)u2.w;
        bfrag[12] = (_Float16)u3.x; bfrag[13] = (_Float16)u3.y;
        bfrag[14] = (_Float16)u3.z; bfrag[15] = (_Float16)u3.w;

        v8f zero = {0.f, 0.f, 0.f, 0.f, 0.f, 0.f, 0.f, 0.f};
        v8f acc[5];
        for (int t = 0; t < 5; ++t) acc[t] = zero;
        for (int t = 0; t < 5; ++t)
            acc[t] = __builtin_amdgcn_wmma_f32_16x16x32_f16(
                false, afrag[t], false, bfrag, (short)0, acc[t], false, false);

        // spill S_in (tiles 1..4) transposed: Re -> cols 0..63, Im -> 64..127
        for (int t = 1; t < 5; ++t) {
            int col = sub * 64 + (t - 1) * 16 + 8 * khalf;
            float4 lo = make_float4(acc[t][0], acc[t][1], acc[t][2], acc[t][3]);
            float4 hi = make_float4(acc[t][4], acc[t][5], acc[t][6], acc[t][7]);
            *(float4*)&S[chunk][col]     = lo;
            *(float4*)&S[chunk][col + 4] = hi;
        }
        __syncthreads();

        // ---- chunk-boundary scan, in place: S_in -> S_state (32 steps) --
        if (tid < N_) {
            int n = tid;
            float lTr = lpow_re[T_][n], lTi = lpow_im[T_][n];
            float sr = 0.f, si = 0.f;
            for (int c = 0; c < C_; ++c) {
                float ir = S[c][n], ii = S[c][64 + n];
                S[c][n] = sr; S[c][64 + n] = si;       // state entering chunk c
                float nr = sr * lTr - si * lTi + ir;   // s' = l^T s + S_in
                float ni = sr * lTi + si * lTr + ii;
                sr = nr; si = ni;
            }
        }
        __syncthreads();

        // ---- phase B: y = y_intra + A2[rows irow] @ S_stack[128 x 32] ---
        v8f yacc = acc[0];                  // intra tile is this wave's C
        for (int q = 0; q < 4; ++q) {
            const float4* sp = (const float4*)&S[chunk][32 * q + 16 * khalf];
            float4 s0 = sp[0], s1 = sp[1], s2 = sp[2], s3 = sp[3];
            v16h bf;
            bf[0]  = (_Float16)s0.x; bf[1]  = (_Float16)s0.y;
            bf[2]  = (_Float16)s0.z; bf[3]  = (_Float16)s0.w;
            bf[4]  = (_Float16)s1.x; bf[5]  = (_Float16)s1.y;
            bf[6]  = (_Float16)s1.z; bf[7]  = (_Float16)s1.w;
            bf[8]  = (_Float16)s2.x; bf[9]  = (_Float16)s2.y;
            bf[10] = (_Float16)s2.z; bf[11] = (_Float16)s2.w;
            bf[12] = (_Float16)s3.x; bf[13] = (_Float16)s3.y;
            bf[14] = (_Float16)s3.z; bf[15] = (_Float16)s3.w;
            yacc = __builtin_amdgcn_wmma_f32_16x16x32_f16(
                false, a2[q], false, bf, (short)0, yacc, false, false);
        }

        // ---- epilogue: skip connection + GELU, 2 x b128 in/out ----------
        {
            const int l0 = chunk * T_ + sub * 16 + 8 * khalf;
            const float4* up4 = (const float4*)(ub + l0);
            float4 uv0 = up4[0], uv1 = up4[1];
            float4 o0, o1;
            o0.x = gelu_tanh(yacc[0] + Dh * uv0.x);
            o0.y = gelu_tanh(yacc[1] + Dh * uv0.y);
            o0.z = gelu_tanh(yacc[2] + Dh * uv0.z);
            o0.w = gelu_tanh(yacc[3] + Dh * uv0.w);
            o1.x = gelu_tanh(yacc[4] + Dh * uv1.x);
            o1.y = gelu_tanh(yacc[5] + Dh * uv1.y);
            o1.z = gelu_tanh(yacc[6] + Dh * uv1.z);
            o1.w = gelu_tanh(yacc[7] + Dh * uv1.w);
            float4* yp4 = (float4*)(y + ((size_t)b * H_ + h) * L_ + l0);
            yp4[0] = o0;
            yp4[1] = o1;
        }
        __syncthreads();   // LDS reused next batch
    }
}

// ---------------------------------------------------------------------------
// pooled[b] = mean_l h[b,:,l];  out[b] = pooled . fc_w + fc_b
// ---------------------------------------------------------------------------
__global__ __launch_bounds__(256)
void pool_fc_kernel(const float* __restrict__ hb, const float* __restrict__ fc_w,
                    const float* __restrict__ fc_b, float* __restrict__ out) {
    __shared__ float red[256];
    int b = blockIdx.x;
    const float* base = hb + (size_t)b * H_ * L_;
    float s = 0.f;
    for (int idx = threadIdx.x; idx < H_ * L_; idx += 256) {
        int h = idx >> 10;                 // / L_
        s += base[idx] * fc_w[h];
    }
    red[threadIdx.x] = s;
    __syncthreads();
    for (int o = 128; o > 0; o >>= 1) {
        if (threadIdx.x < o) red[threadIdx.x] += red[threadIdx.x + o];
        __syncthreads();
    }
    if (threadIdx.x == 0) out[b] = red[0] / (float)L_ + fc_b[0];
}

// ---------------------------------------------------------------------------
extern "C" void kernel_launch(void* const* d_in, const int* in_sizes, int n_in,
                              void* d_out, int out_size, void* d_ws, size_t ws_size,
                              hipStream_t stream) {
    const int*   x    = (const int*)d_in[0];
    const float* emb  = (const float*)d_in[1];
    const float* ldt  = (const float*)d_in[2];
    const float* Alog = (const float*)d_in[3];
    const float* Aim  = (const float*)d_in[4];
    const float* Cre  = (const float*)d_in[5];
    const float* Cim  = (const float*)d_in[6];
    const float* Dp   = (const float*)d_in[7];
    const float* fcw  = (const float*)d_in[8];
    const float* fcb  = (const float*)d_in[9];
    float* out  = (float*)d_out;
    float* buf0 = (float*)d_ws;
    float* buf1 = buf0 + BHL;

    embed_kernel<<<B_ * (L_ / 32), 256, 0, stream>>>(x, emb, buf0);

    float* cur = buf0;
    float* nxt = buf1;
    for (int i = 0; i < NL_; ++i) {
        dim3 grid(H_, 4);
        s4d_kernel<<<grid, 128, 0, stream>>>(
            cur, nxt,
            ldt + (size_t)i * H_,
            Alog + (size_t)i * H_ * N_, Aim + (size_t)i * H_ * N_,
            Cre + (size_t)i * H_ * N_, Cim + (size_t)i * H_ * N_,
            Dp + (size_t)i * H_);
        float* t = cur; cur = nxt; nxt = t;
    }

    pool_fc_kernel<<<B_, 256, 0, stream>>>(cur, fcw, fcb, out);
}
